// TransformerNet_38036230374148
// MI455X (gfx1250) — matrix operand order repaired and checked by reading, not compiled, not run on hardware
//
#include <hip/hip_runtime.h>
#include <math.h>

// ---------------------------------------------------------------------------
// Types / helpers
// ---------------------------------------------------------------------------
typedef __attribute__((ext_vector_type(16))) __bf16 v16bf;
typedef __attribute__((ext_vector_type(8)))  float  v8f;
typedef __attribute__((ext_vector_type(4)))  int    v4i;

#define DEV __device__ __forceinline__

// Probe for gfx1250 async global->LDS builtins (compile-safe fallback).
#if defined(__has_builtin)
# if __has_builtin(__builtin_amdgcn_global_load_async_to_lds_b128) && \
     __has_builtin(__builtin_amdgcn_s_wait_asynccnt)
#  define HAVE_ASYNC 1
# endif
#endif
#ifndef HAVE_ASYNC
# define HAVE_ASYNC 0
#endif

DEV unsigned short f2bf(float f) {             // round-to-nearest-even f32->bf16
  union { float f; unsigned u; } v; v.f = f;
  unsigned r = v.u + 0x7fffu + ((v.u >> 16) & 1u);
  return (unsigned short)(r >> 16);
}
DEV float bf2f(unsigned short u) {
  union { unsigned u; float f; } v; v.u = ((unsigned)u) << 16;
  return v.f;
}
DEV v8f vzero() { v8f z = {0.f,0.f,0.f,0.f,0.f,0.f,0.f,0.f}; return z; }

union Frag16 { uint4 q[2]; v16bf bf; };

// A fragment, 16x32 bf16 (MxK).  Layout: lanes 0-15 rows 0-15 hold
// K 0..15 (contiguous); lanes 16-31 hold K 16..31.
DEV v16bf load_a(const unsigned short* base, int ld, int row0, int k0, int lane) {
  int lr = lane & 15, lh = lane >> 4;
  const unsigned short* p = base + (size_t)(row0 + lr) * ld + k0 + lh * 16;
  Frag16 f;
  f.q[0] = *(const uint4*)p;          // K = k0 + lh*16 .. +8
  f.q[1] = *(const uint4*)(p + 8);    // K = k0 + lh*16 + 8 .. +8
  return f.bf;
}
// B fragment, 32x16 bf16 (KxN) sourced from row-major W[N,K]:
// lane l<16 holds column n=l, K 0..15 (contiguous); lanes 16-31 K 16..31.
DEV v16bf load_b(const unsigned short* base, int ld, int n0, int k0, int lane) {
  int lr = lane & 15, lh = lane >> 4;
  const unsigned short* p = base + (size_t)(n0 + lr) * ld + k0 + lh * 16;
  Frag16 f;
  f.q[0] = *(const uint4*)p;
  f.q[1] = *(const uint4*)(p + 8);
  return f.bf;
}
DEV v8f wmma_bf16(v16bf a, v16bf b, v8f c) {
  return __builtin_amdgcn_wmma_f32_16x16x32_bf16(false, a, false, b,
                                                 (short)0, c, false, false);
}

#if HAVE_ASYNC
DEV void async_copy_b128(const unsigned short* g, unsigned short* l) {
  __builtin_amdgcn_global_load_async_to_lds_b128(
      (__attribute__((address_space(1))) v4i*)(g),
      (__attribute__((address_space(3))) v4i*)(l), 0, 0);
}
#endif

// ---------------------------------------------------------------------------
// f32 -> bf16 convert
// ---------------------------------------------------------------------------
__global__ void f32_to_bf16(const float* __restrict__ in,
                            unsigned short* __restrict__ out, long n) {
  for (long i = (long)blockIdx.x * blockDim.x + threadIdx.x; i < n;
       i += (long)gridDim.x * blockDim.x)
    out[i] = f2bf(in[i]);
}

// ---------------------------------------------------------------------------
// WMMA GEMM:  C[M,N] = A[M,K] @ W[N,K]^T + bias  (+res, LN, relu)
// Wave grid WRM x WRN (8 waves); each wave computes a 32x64 strip
// (2 A-frags x 4 B-frags -> 8 WMMAs per K-step).  A tile is staged through
// LDS (double buffered, async copy when available); B comes straight from
// global (weights are L2 resident) — addressed off the kernarg pointer so
// the compiler keeps it in the global (not flat) address space.
// LN epilogue requires BN == N and gridDim.x == 1.
// ---------------------------------------------------------------------------
template<int WRM, int WRN, bool RES, bool LN, bool RELU, bool OUTF32>
__global__ void __launch_bounds__(256) wgemm(
    const unsigned short* __restrict__ A,
    const unsigned short* __restrict__ W,
    const float* __restrict__ bias,
    const unsigned short* __restrict__ Rsd,
    const float* __restrict__ gamma,
    const float* __restrict__ beta,
    void* __restrict__ Out,
    int M, int N, int K, int ldc)
{
  constexpr int BM = WRM * 32;
  constexpr int BN = WRN * 64;
  static_assert(WRM * WRN == 8, "block must be 8 waves");
  __shared__ unsigned short sA[2][BM * 32];
  __shared__ float sC[LN ? (BM * BN) : 1];

  const int tid  = threadIdx.x;
  const int wave = tid >> 5;
  const int lane = tid & 31;
  const int wm = wave / WRN, wn = wave % WRN;
  const int mblk = blockIdx.y * BM;
  const int m0 = wm * 32;                       // wave row (block-local)
  const int n0 = blockIdx.x * BN + wn * 64;     // wave col (global)
  const int lr = lane & 15, lh = lane >> 4;

  // B row element offsets (kept as integer offsets off W so loads stay global)
  size_t boff[4];
  #pragma unroll
  for (int t = 0; t < 4; ++t) {
    int c = n0 + t * 16 + lr;
    if (c > N - 1) c = N - 1;
    boff[t] = (size_t)c * K + (size_t)(lh * 16);
  }

  v8f acc[2][4];
  #pragma unroll
  for (int i = 0; i < 2; ++i)
    #pragma unroll
    for (int t = 0; t < 4; ++t) acc[i][t] = vzero();

  auto copyA = [&](int k0, int buf) {
    for (int c = tid; c < BM * 4; c += 256) {
      int row = c >> 2, ch = c & 3;
      const unsigned short* g = A + (size_t)(mblk + row) * K + k0 + ch * 8;
#if HAVE_ASYNC
      async_copy_b128(g, &sA[buf][row * 32 + ch * 8]);
#else
      uint4 d = *(const uint4*)g;
      *(uint4*)&sA[buf][row * 32 + ch * 8] = d;
#endif
    }
  };

  copyA(0, 0);
  int cur = 0;
  for (int k0 = 0; k0 < K; k0 += 32) {
#if HAVE_ASYNC
    __builtin_amdgcn_s_wait_asynccnt(0);
#endif
    __syncthreads();                       // tile[cur] visible to all waves
    if (k0 + 32 < K) {
      copyA(k0 + 32, cur ^ 1);
      #pragma unroll
      for (int t = 0; t < 4; ++t)
        __builtin_prefetch(W + boff[t] + k0 + 32, 0, 1);
    }
    v16bf a0 = load_a(sA[cur], 32, m0,      0, lane);
    v16bf a1 = load_a(sA[cur], 32, m0 + 16, 0, lane);
    Frag16 bf[4];
    #pragma unroll
    for (int t = 0; t < 4; ++t) {
      const unsigned short* p = W + boff[t] + k0;
      bf[t].q[0] = *(const uint4*)p;
      bf[t].q[1] = *(const uint4*)(p + 8);
    }
    #pragma unroll
    for (int t = 0; t < 4; ++t) {
      acc[0][t] = wmma_bf16(a0, bf[t].bf, acc[0][t]);
      acc[1][t] = wmma_bf16(a1, bf[t].bf, acc[1][t]);
    }
    __syncthreads();                       // reads of tile[cur] complete
    cur ^= 1;
  }

  if (!LN) {
    #pragma unroll
    for (int i = 0; i < 2; ++i)
      #pragma unroll
      for (int r = 0; r < 8; ++r) {
        int row = mblk + m0 + i * 16 + lh * 8 + r;
        if (row >= M) continue;
        #pragma unroll
        for (int t = 0; t < 4; ++t) {
          int col = n0 + t * 16 + lr;
          if (col >= N) continue;
          float v = acc[i][t][r] + bias[col];
          if (RES)  v += bf2f(Rsd[(size_t)row * N + col]);
          if (RELU) v = fmaxf(v, 0.f);
          if (OUTF32) ((float*)Out)[(size_t)row * ldc + col] = v;
          else ((unsigned short*)Out)[(size_t)row * ldc + col] = f2bf(v);
        }
      }
  } else {
    // stage (bias + residual) into LDS, then per-row LayerNorm
    #pragma unroll
    for (int i = 0; i < 2; ++i)
      #pragma unroll
      for (int r = 0; r < 8; ++r) {
        int rl = m0 + i * 16 + lh * 8 + r;
        #pragma unroll
        for (int t = 0; t < 4; ++t) {
          int cl = wn * 64 + t * 16 + lr;        // BN == N, blockIdx.x == 0
          float v = acc[i][t][r] + bias[cl];
          if (RES) v += bf2f(Rsd[(size_t)(mblk + rl) * N + cl]);
          sC[rl * BN + cl] = v;
        }
      }
    __syncthreads();
    constexpr int RPW = BM / 8;
    for (int rr = 0; rr < RPW; ++rr) {
      int rl  = wave * RPW + rr;
      int row = mblk + rl;
      float s = 0.f, s2 = 0.f;
      for (int c = lane; c < BN; c += 32) {
        float v = sC[rl * BN + c]; s += v; s2 += v * v;
      }
      for (int off = 16; off; off >>= 1) {
        s  += __shfl_xor(s,  off, 32);
        s2 += __shfl_xor(s2, off, 32);
      }
      float mean = s / BN;
      float rstd = rsqrtf(s2 / BN - mean * mean + 1e-5f);
      for (int c = lane; c < BN; c += 32) {
        float v = (sC[rl * BN + c] - mean) * rstd * gamma[c] + beta[c];
        if (RELU) v = fmaxf(v, 0.f);
        if (OUTF32) ((float*)Out)[(size_t)row * ldc + c] = v;
        else ((unsigned short*)Out)[(size_t)row * ldc + c] = f2bf(v);
      }
    }
  }
}

// ---------------------------------------------------------------------------
// Fused attention per (b,h): [optional pe: LN(pe2 slice) @ pdec_w^T + pdec_b]
//   att = softmax(Q K^T * scale + pe);  O = att @ V
// S=64, HD=32.  4 waves; wave w owns att rows [16w,16w+16).
// ---------------------------------------------------------------------------
template<bool PE>
__global__ void __launch_bounds__(128) attn_kernel(
    const unsigned short* __restrict__ Qb, int q_ld, int q_bstr, int q_hoff,
    const unsigned short* __restrict__ Kb, int k_ld, int k_bstr, int k_hoff,
    const unsigned short* __restrict__ Vb, int v_ld, int v_bstr, int v_hoff,
    const unsigned short* __restrict__ pe2,
    const float* __restrict__ g2, const float* __restrict__ be2,
    const unsigned short* __restrict__ pdw, const float* __restrict__ pdb,
    unsigned short* __restrict__ Ob, int o_ld, int o_bstr, int o_hoff,
    float scale)
{
  const int h = blockIdx.x, b = blockIdx.y;
  const int wave = threadIdx.x >> 5, lane = threadIdx.x & 31;
  const int lr = lane & 15, lh = lane >> 4;

  __shared__ unsigned short sAtt[64 * 64];
  __shared__ unsigned short sVt [32 * 64];
  __shared__ unsigned short sPe [64 * 64];

  const unsigned short* q = Qb + (size_t)b * q_bstr + (size_t)h * q_hoff;
  const unsigned short* k = Kb + (size_t)b * k_bstr + (size_t)h * k_hoff;
  const unsigned short* v = Vb + (size_t)b * v_bstr + (size_t)h * v_hoff;

  // stage V^T into LDS: sVt[d][kk]
  for (int i = threadIdx.x; i < 32 * 64; i += 128) {
    int d = i & 31, kk = i >> 5;
    sVt[d * 64 + kk] = v[(size_t)kk * v_ld + d];
  }

  v8f pacc[4] = {vzero(), vzero(), vzero(), vzero()};
  if (PE) {
    if (threadIdx.x < 64) {                   // per-row LN of pe2 slice
      int s = threadIdx.x;
      const unsigned short* pr = pe2 + ((size_t)(b * 64 + s)) * 512 + h * 64;
      float vals[64]; float su = 0.f, sq = 0.f;
      for (int j = 0; j < 64; ++j) { float f = bf2f(pr[j]); vals[j] = f; su += f; sq += f * f; }
      float mean = su * (1.f / 64.f);
      float rstd = rsqrtf(sq * (1.f / 64.f) - mean * mean + 1e-5f);
      for (int j = 0; j < 64; ++j)
        sPe[s * 64 + j] = f2bf((vals[j] - mean) * rstd * g2[j] + be2[j]);
    }
    __syncthreads();
    for (int k0 = 0; k0 < 64; k0 += 32) {     // pe-bias GEMM 64x64x64
      v16bf a = load_a(sPe, 64, wave * 16, k0, lane);
      #pragma unroll
      for (int t = 0; t < 4; ++t) {
        v16bf bb = load_b(pdw, 64, t * 16, k0, lane);
        pacc[t] = wmma_bf16(a, bb, pacc[t]);
      }
    }
  } else {
    __syncthreads();
  }

  // Q K^T (single K step, HD=32)
  v8f qk[4] = {vzero(), vzero(), vzero(), vzero()};
  {
    v16bf a = load_a(q, q_ld, wave * 16, 0, lane);
    #pragma unroll
    for (int t = 0; t < 4; ++t) {
      v16bf bb = load_b(k, k_ld, t * 16, 0, lane);
      qk[t] = wmma_bf16(a, bb, qk[t]);
    }
  }

  // softmax over the wave's 16 rows (rows split across lane halves)
  float pv[4][8], rmax[8], rsum[8];
  #pragma unroll
  for (int r = 0; r < 8; ++r) rmax[r] = -1e30f;
  #pragma unroll
  for (int t = 0; t < 4; ++t)
    #pragma unroll
    for (int r = 0; r < 8; ++r) {
      float val = qk[t][r] * scale;
      if (PE) val += pacc[t][r] + pdb[t * 16 + lr];
      pv[t][r] = val;
      rmax[r] = fmaxf(rmax[r], val);
    }
  #pragma unroll
  for (int r = 0; r < 8; ++r) {
    for (int off = 8; off; off >>= 1)
      rmax[r] = fmaxf(rmax[r], __shfl_xor(rmax[r], off, 32));
    rsum[r] = 0.f;
  }
  #pragma unroll
  for (int t = 0; t < 4; ++t)
    #pragma unroll
    for (int r = 0; r < 8; ++r) {
      float e = __expf(pv[t][r] - rmax[r]);
      pv[t][r] = e; rsum[r] += e;
    }
  #pragma unroll
  for (int r = 0; r < 8; ++r)
    for (int off = 8; off; off >>= 1) rsum[r] += __shfl_xor(rsum[r], off, 32);

  #pragma unroll
  for (int t = 0; t < 4; ++t)
    #pragma unroll
    for (int r = 0; r < 8; ++r) {
      int row = wave * 16 + lh * 8 + r;
      int col = t * 16 + lr;
      sAtt[row * 64 + col] = f2bf(pv[t][r] / rsum[r]);
    }
  __syncthreads();

  // O = att @ V  (16x32 strip per wave, K=64)
  v8f oacc[2] = {vzero(), vzero()};
  for (int k0 = 0; k0 < 64; k0 += 32) {
    v16bf a = load_a(sAtt, 64, wave * 16, k0, lane);
    #pragma unroll
    for (int t = 0; t < 2; ++t) {
      v16bf bb = load_b(sVt, 64, t * 16, k0, lane);
      oacc[t] = wmma_bf16(a, bb, oacc[t]);
    }
  }
  unsigned short* ob = Ob + (size_t)b * o_bstr + (size_t)h * o_hoff;
  #pragma unroll
  for (int t = 0; t < 2; ++t)
    #pragma unroll
    for (int r = 0; r < 8; ++r) {
      int row = wave * 16 + lh * 8 + r;
      int col = t * 16 + lr;
      ob[(size_t)row * o_ld + col] = f2bf(oacc[t][r]);
    }
}

// ---------------------------------------------------------------------------
// Input embedding: e = x@emb_w^T+b; h = relu(LN([x,e])); X1 = [h, amap] bf16
// One wave per token, 8 tokens / block.
// ---------------------------------------------------------------------------
__global__ void __launch_bounds__(256) embed_kernel(
    const float* __restrict__ x, const float* __restrict__ amap,
    const float* __restrict__ emb_w, const float* __restrict__ emb_b,
    const float* __restrict__ g1, const float* __restrict__ b1,
    unsigned short* __restrict__ X1)
{
  __shared__ float sH[8][128];
  const int wave = threadIdx.x >> 5, lane = threadIdx.x & 31;
  const int tok = blockIdx.x * 8 + wave;
  const int s = tok & 63;
  const float* xr = x + (size_t)tok * 80;

  for (int i = lane; i < 80; i += 32) sH[wave][i] = xr[i];
  __syncthreads();
  for (int o = lane; o < 48; o += 32) {
    float acc = emb_b[o];
    const float* wr = emb_w + o * 80;
    for (int kk = 0; kk < 80; ++kk) acc += sH[wave][kk] * wr[kk];
    sH[wave][80 + o] = acc;
  }
  __syncthreads();
  float lsum = 0.f, lsq = 0.f, vals[4];
  #pragma unroll
  for (int j = 0; j < 4; ++j) {
    float f = sH[wave][lane * 4 + j];
    vals[j] = f; lsum += f; lsq += f * f;
  }
  for (int off = 16; off; off >>= 1) {
    lsum += __shfl_xor(lsum, off, 32);
    lsq  += __shfl_xor(lsq,  off, 32);
  }
  float mean = lsum * (1.f / 128.f);
  float rstd = rsqrtf(lsq * (1.f / 128.f) - mean * mean + 1e-5f);
  unsigned short* orow = X1 + (size_t)tok * 192;
  #pragma unroll
  for (int j = 0; j < 4; ++j) {
    int c = lane * 4 + j;
    float vv = (vals[j] - mean) * rstd * g1[c] + b1[c];
    orow[c] = f2bf(fmaxf(vv, 0.f));
  }
  #pragma unroll
  for (int j = 0; j < 2; ++j) {
    int c = lane * 2 + j;
    orow[128 + c] = f2bf(amap[s * 64 + c]);
  }
}

// ---------------------------------------------------------------------------
// value = tanh(v2 @ w3^T + b3)
// ---------------------------------------------------------------------------
__global__ void value_tail(const unsigned short* __restrict__ v2,
                           const float* __restrict__ w3,
                           const float* __restrict__ b3,
                           float* __restrict__ out)
{
  int b = blockIdx.x * 256 + threadIdx.x;
  if (b >= 1024) return;
  float acc = b3[0];
  for (int kk = 0; kk < 64; ++kk) acc += bf2f(v2[b * 64 + kk]) * w3[kk];
  out[b] = tanhf(acc);
}

// ---------------------------------------------------------------------------
// Host orchestration
// ---------------------------------------------------------------------------
extern "C" void kernel_launch(void* const* d_in, const int* in_sizes, int n_in,
                              void* d_out, int out_size, void* d_ws, size_t ws_size,
                              hipStream_t stream)
{
  (void)in_sizes; (void)n_in; (void)out_size; (void)ws_size;
  const float* x       = (const float*)d_in[0];
  const float* amap    = (const float*)d_in[1];
  const float* emb_w   = (const float*)d_in[2];
  const float* emb_b   = (const float*)d_in[3];
  const float* e_ln1g  = (const float*)d_in[4];
  const float* e_ln1b  = (const float*)d_in[5];
  const float* lin1_w  = (const float*)d_in[6];
  const float* lin1_b  = (const float*)d_in[7];
  const float* e_ln2g  = (const float*)d_in[8];
  const float* e_ln2b  = (const float*)d_in[9];
  const float* qkv_w   = (const float*)d_in[10];
  const float* qkv_b   = (const float*)d_in[11];
  const float* out_w   = (const float*)d_in[12];
  const float* out_b   = (const float*)d_in[13];
  const float* penc_w  = (const float*)d_in[14];
  const float* penc_b  = (const float*)d_in[15];
  const float* penc_g1 = (const float*)d_in[16];
  const float* penc_b1 = (const float*)d_in[17];
  const float* plin_w  = (const float*)d_in[18];
  const float* plin_b  = (const float*)d_in[19];
  const float* penc_g2 = (const float*)d_in[20];
  const float* penc_b2 = (const float*)d_in[21];
  const float* pdec_w  = (const float*)d_in[22];
  const float* pdec_b  = (const float*)d_in[23];
  const float* l_ln1g  = (const float*)d_in[24];
  const float* l_ln1b  = (const float*)d_in[25];
  const float* ff1_w   = (const float*)d_in[26];
  const float* ff1_b   = (const float*)d_in[27];
  const float* ff2_w   = (const float*)d_in[28];
  const float* ff2_b   = (const float*)d_in[29];
  const float* l_ln2g  = (const float*)d_in[30];
  const float* l_ln2b  = (const float*)d_in[31];
  const float* kv_w    = (const float*)d_in[32];
  const float* kv_b    = (const float*)d_in[33];
  const float* q_w     = (const float*)d_in[34];
  const float* q_b     = (const float*)d_in[35];
  const float* pout_w  = (const float*)d_in[36];
  const float* pout_b  = (const float*)d_in[37];
  const float* p_lng   = (const float*)d_in[38];
  const float* p_lnb   = (const float*)d_in[39];
  const float* pol_w   = (const float*)d_in[40];
  const float* pol_b   = (const float*)d_in[41];
  const float* v_w1    = (const float*)d_in[42];
  const float* v_b1    = (const float*)d_in[43];
  const float* v_w2    = (const float*)d_in[44];
  const float* v_b2    = (const float*)d_in[45];
  const float* v_w3    = (const float*)d_in[46];
  const float* v_b3    = (const float*)d_in[47];

  char* base = (char*)d_ws;
  size_t off = 0;
  auto alloc_us = [&](size_t n) -> unsigned short* {
    unsigned short* p = (unsigned short*)(base + off);
    off = (off + n * 2 + 255) & ~(size_t)255;
    return p;
  };

  const int T = 1024 * 64;

  // bf16 weight copies
  unsigned short* amap_h  = alloc_us(64 * 64);
  unsigned short* lin1_h  = alloc_us(256 * 192);
  unsigned short* qkv_h   = alloc_us((size_t)8 * 768 * 256);
  unsigned short* outw_h  = alloc_us((size_t)8 * 256 * 256);
  unsigned short* pencw_h = alloc_us((size_t)8 * 64 * 256);
  unsigned short* plinw_h = alloc_us((size_t)8 * 512 * 64);
  unsigned short* pdecw_h = alloc_us((size_t)8 * 64 * 64);
  unsigned short* ff1w_h  = alloc_us((size_t)8 * 1024 * 256);
  unsigned short* ff2w_h  = alloc_us((size_t)8 * 256 * 1024);
  unsigned short* kvw_h   = alloc_us((size_t)512 * 256);
  unsigned short* qw_h    = alloc_us((size_t)256 * 64);
  unsigned short* poutw_h = alloc_us((size_t)256 * 256);
  unsigned short* polw_h  = alloc_us((size_t)73 * 256);
  unsigned short* vw1_h   = alloc_us((size_t)256 * 256);
  unsigned short* vw2_h   = alloc_us((size_t)64 * 16384);

  // activations (BIG reused for qkv/ff1/v1; PE2 reused for kv; OB for both attns)
  unsigned short* X1  = alloc_us((size_t)T * 192);
  unsigned short* Hb  = alloc_us((size_t)T * 256);
  unsigned short* H2  = alloc_us((size_t)T * 256);
  unsigned short* BIG = alloc_us((size_t)T * 1024);
  unsigned short* PE2 = alloc_us((size_t)T * 512);
  unsigned short* OB  = alloc_us((size_t)T * 256);
  unsigned short* PEN = alloc_us((size_t)T * 64);
  unsigned short* QB  = alloc_us((size_t)64 * 256);
  unsigned short* V2  = alloc_us((size_t)1024 * 64);

  auto cvt = [&](const float* src, unsigned short* dst, size_t n) {
    unsigned blocks = (unsigned)((n + 2047) / 2048);
    if (blocks > 4096) blocks = 4096;
    f32_to_bf16<<<blocks, 256, 0, stream>>>(src, dst, (long)n);
  };
  cvt(amap,   amap_h,  64 * 64);
  cvt(lin1_w, lin1_h,  256 * 192);
  cvt(qkv_w,  qkv_h,   (size_t)8 * 768 * 256);
  cvt(out_w,  outw_h,  (size_t)8 * 256 * 256);
  cvt(penc_w, pencw_h, (size_t)8 * 64 * 256);
  cvt(plin_w, plinw_h, (size_t)8 * 512 * 64);
  cvt(pdec_w, pdecw_h, (size_t)8 * 64 * 64);
  cvt(ff1_w,  ff1w_h,  (size_t)8 * 1024 * 256);
  cvt(ff2_w,  ff2w_h,  (size_t)8 * 256 * 1024);
  cvt(kv_w,   kvw_h,   512 * 256);
  cvt(q_w,    qw_h,    256 * 64);
  cvt(pout_w, poutw_h, 256 * 256);
  cvt(pol_w,  polw_h,  73 * 256);
  cvt(v_w1,   vw1_h,   256 * 256);
  cvt(v_w2,   vw2_h,   (size_t)64 * 16384);

  // ---- input embedding ----
  embed_kernel<<<T / 8, 256, 0, stream>>>(x, amap, emb_w, emb_b, e_ln1g, e_ln1b, X1);
  wgemm<2, 4, false, true, true, false><<<dim3(1, T / 64), 256, 0, stream>>>(
      X1, lin1_h, lin1_b, nullptr, e_ln2g, e_ln2b, Hb, T, 256, 192, 256);

  const float scale = 0.1767766952966369f;  // 1/sqrt(32)

  // ---- transformer layers ----
  for (int l = 0; l < 8; ++l) {
    const unsigned short* qkvw  = qkv_h   + (size_t)l * 768 * 256;
    const unsigned short* outw  = outw_h  + (size_t)l * 256 * 256;
    const unsigned short* pencw = pencw_h + (size_t)l * 64 * 256;
    const unsigned short* plinw = plinw_h + (size_t)l * 512 * 64;
    const unsigned short* pdecw = pdecw_h + (size_t)l * 64 * 64;
    const unsigned short* ff1w  = ff1w_h  + (size_t)l * 1024 * 256;
    const unsigned short* ff2w  = ff2w_h  + (size_t)l * 256 * 1024;

    wgemm<2, 4, false, false, false, false><<<dim3(3, T / 64), 256, 0, stream>>>(
        Hb, qkvw, qkv_b + l * 768, nullptr, nullptr, nullptr, BIG, T, 768, 256, 768);
    wgemm<8, 1, false, true, false, false><<<dim3(1, T / 256), 256, 0, stream>>>(
        Hb, pencw, penc_b + l * 64, nullptr, penc_g1 + l * 64, penc_b1 + l * 64,
        PEN, T, 64, 256, 64);
    wgemm<2, 4, false, false, false, false><<<dim3(2, T / 64), 256, 0, stream>>>(
        PEN, plinw, plin_b + l * 512, nullptr, nullptr, nullptr, PE2, T, 512, 64, 512);
    attn_kernel<true><<<dim3(8, 1024), 128, 0, stream>>>(
        BIG,      768, 64 * 768, 96,
        BIG + 32, 768, 64 * 768, 96,
        BIG + 64, 768, 64 * 768, 96,
        PE2, penc_g2 + l * 64, penc_b2 + l * 64,
        pdecw, pdec_b + l * 64,
        OB, 256, 64 * 256, 32, scale);
    wgemm<2, 4, true, true, false, false><<<dim3(1, T / 64), 256, 0, stream>>>(
        OB, outw, out_b + l * 256, Hb, l_ln1g + l * 256, l_ln1b + l * 256,
        H2, T, 256, 256, 256);
    wgemm<2, 4, false, false, true, false><<<dim3(4, T / 64), 256, 0, stream>>>(
        H2, ff1w, ff1_b + l * 1024, nullptr, nullptr, nullptr, BIG, T, 1024, 256, 1024);
    wgemm<2, 4, true, true, false, false><<<dim3(1, T / 64), 256, 0, stream>>>(
        BIG, ff2w, ff2_b + l * 256, H2, l_ln2g + l * 256, l_ln2b + l * 256,
        Hb, T, 256, 1024, 256);
  }

  // ---- value head ----
  wgemm<2, 4, false, false, true, false><<<dim3(1, T / 64), 256, 0, stream>>>(
      Hb, vw1_h, v_b1, nullptr, nullptr, nullptr, BIG, T, 256, 256, 256);
  wgemm<8, 1, false, false, true, false><<<dim3(1, 1024 / 256), 256, 0, stream>>>(
      BIG, vw2_h, v_b2, nullptr, nullptr, nullptr, V2, 1024, 64, 16384, 64);
  value_tail<<<4, 256, 0, stream>>>(V2, v_w3, v_b3,
                                    (float*)d_out + (size_t)1024 * 64 * 73);

  // ---- policy head ----
  wgemm<2, 4, false, false, false, false><<<dim3(2, T / 64), 256, 0, stream>>>(
      Hb, kvw_h, kv_b, nullptr, nullptr, nullptr, PE2, T, 512, 256, 512);
  wgemm<2, 4, false, false, false, false><<<dim3(1, 1), 256, 0, stream>>>(
      amap_h, qw_h, q_b, nullptr, nullptr, nullptr, QB, 64, 256, 64, 256);
  attn_kernel<false><<<dim3(8, 1024), 128, 0, stream>>>(
      QB,       256, 0,        32,
      PE2,      512, 64 * 512, 64,
      PE2 + 32, 512, 64 * 512, 64,
      nullptr, nullptr, nullptr, nullptr, nullptr,
      OB, 256, 64 * 256, 32, scale);
  wgemm<2, 4, true, true, true, false><<<dim3(1, T / 64), 256, 0, stream>>>(
      OB, poutw_h, pout_b, Hb, p_lng, p_lnb, H2, T, 256, 256, 256);
  wgemm<2, 4, false, false, false, true><<<dim3(1, T / 64), 256, 0, stream>>>(
      H2, polw_h, pol_b, nullptr, nullptr, nullptr, (float*)d_out, T, 73, 256, 73);
}
